// Local_MP_32057635897562
// MI455X (gfx1250) — compile-verified, matrix operand order checked by reading, not run on hardware
//
#include <hip/hip_runtime.h>
#include <hip/hip_bf16.h>
#include <cstdint>
#include <cstddef>

#define DIM    256
#define NNODES 16384
#define NEDGES 200000
#define NTRI   400000

typedef __attribute__((ext_vector_type(16))) __bf16 v16bf;
typedef __attribute__((ext_vector_type(8)))  float  v8f;

__device__ __forceinline__ float silu_f(float x) { return x / (1.0f + __expf(-x)); }

__device__ __forceinline__ void atomic_add_f32(float* p, float v) {
    unsafeAtomicAdd(p, v);   // native global_atomic_add_f32 path
}

// ---------------- weight prep: W[K,256] fp32 (row-major) -> Wt[256,K] bf16 ----------------
__global__ void prep_w_kernel(const float* __restrict__ W, __bf16* __restrict__ Wt, int K) {
    int o = blockIdx.x * 256 + threadIdx.x;     // o in [0, 256*K)
    int n = o / K;
    int k = o - n * K;
    Wt[o] = (__bf16)W[(size_t)k * DIM + n];
}

// ---------------- fused GEMM: out = (maybe_silu(X@W + b) + add) * mul, N = 256 ----------------
// Block: 512 threads = 16 waves; one 16-row tile per block, one 16-col slab per wave.
template<int K, bool SILU>
__global__ __launch_bounds__(512) void gemm_bf16_kernel(
    const float* __restrict__ X, const __bf16* __restrict__ Wt,
    const float* __restrict__ bias, const float* __restrict__ addv,
    const float* __restrict__ mulv, float* __restrict__ out)
{
    __shared__ __attribute__((aligned(32))) __bf16 As[16 * K];
    const int row0 = blockIdx.x * 16;
    const int tid  = threadIdx.x;

    // cooperative stage: 16*K fp32 -> bf16 in LDS (coalesced b128 global loads)
    const float4* src = (const float4*)(X + (size_t)row0 * K);
    constexpr int NV = (16 * K / 4) / 512;      // 2 (K=256) or 6 (K=768)
    #pragma unroll
    for (int it = 0; it < NV; ++it) {
        int v = tid + it * 512;
        float4 d = src[v];
        As[v * 4 + 0] = (__bf16)d.x;
        As[v * 4 + 1] = (__bf16)d.y;
        As[v * 4 + 2] = (__bf16)d.z;
        As[v * 4 + 3] = (__bf16)d.w;
    }
    __syncthreads();

    const int lane  = tid & 31;
    const int wave  = tid >> 5;                 // 0..15 -> n-tile
    const int mrow  = lane & 15;
    const int khalf = lane >> 4;
    const int ncol  = (wave << 4) + (lane & 15);

    v8f acc = {0.f, 0.f, 0.f, 0.f, 0.f, 0.f, 0.f, 0.f};
    const __bf16* wcol = Wt + (size_t)ncol * K;
    #pragma unroll 4
    for (int kc = 0; kc < K / 32; ++kc) {
        v16bf a = *(const v16bf*)&As[mrow * K + kc * 32 + khalf * 16];
        v16bf b = *(const v16bf*)&wcol[kc * 32 + khalf * 16];
        acc = __builtin_amdgcn_wmma_f32_16x16x32_bf16(false, a, false, b, (short)0, acc, false, false);
    }

    const float bc = bias ? bias[ncol] : 0.0f;
    #pragma unroll
    for (int r = 0; r < 8; ++r) {
        const int m = r + khalf * 8;
        const size_t oi = (size_t)(row0 + m) * DIM + ncol;
        float v = acc[r] + bc;
        if (SILU) v = silu_f(v);
        if (addv) v += addv[oi];
        if (mulv) v *= mulv[oi];
        out[oi] = v;
    }
}

// ------- fused: sbfmlp = silu(silu(sbf@W0+b0)@W1+b1); seg[idx_s] += sbfmlp * m_pre[idx_g] -------
__global__ __launch_bounds__(512) void sbf_mlp_scatter_kernel(
    const float* __restrict__ sbf,
    const __bf16* __restrict__ Wt0, const float* __restrict__ b0,
    const __bf16* __restrict__ Wt1, const float* __restrict__ b1,
    const float* __restrict__ m_pre, const int* __restrict__ idx_g,
    const int* __restrict__ idx_s, float* __restrict__ seg)
{
    __shared__ __attribute__((aligned(32))) __bf16 As[16 * DIM];
    __shared__ __attribute__((aligned(32))) __bf16 Hs[16 * DIM];
    const int row0 = blockIdx.x * 16;
    const int tid  = threadIdx.x;

    const float4* src = (const float4*)(sbf + (size_t)row0 * DIM);
    #pragma unroll
    for (int it = 0; it < 2; ++it) {
        int v = tid + it * 512;
        float4 d = src[v];
        As[v * 4 + 0] = (__bf16)d.x;
        As[v * 4 + 1] = (__bf16)d.y;
        As[v * 4 + 2] = (__bf16)d.z;
        As[v * 4 + 3] = (__bf16)d.w;
    }
    __syncthreads();

    const int lane  = tid & 31;
    const int wave  = tid >> 5;
    const int mrow  = lane & 15;
    const int khalf = lane >> 4;
    const int ncol  = (wave << 4) + (lane & 15);

    // layer 1
    v8f acc = {0.f, 0.f, 0.f, 0.f, 0.f, 0.f, 0.f, 0.f};
    const __bf16* w0 = Wt0 + (size_t)ncol * DIM;
    #pragma unroll
    for (int kc = 0; kc < DIM / 32; ++kc) {
        v16bf a = *(const v16bf*)&As[mrow * DIM + kc * 32 + khalf * 16];
        v16bf b = *(const v16bf*)&w0[kc * 32 + khalf * 16];
        acc = __builtin_amdgcn_wmma_f32_16x16x32_bf16(false, a, false, b, (short)0, acc, false, false);
    }
    const float bc0 = b0[ncol];
    #pragma unroll
    for (int r = 0; r < 8; ++r) {
        const int m = r + khalf * 8;
        Hs[m * DIM + ncol] = (__bf16)silu_f(acc[r] + bc0);
    }
    __syncthreads();

    // layer 2 + gather/mul/scatter
    v8f acc2 = {0.f, 0.f, 0.f, 0.f, 0.f, 0.f, 0.f, 0.f};
    const __bf16* w1 = Wt1 + (size_t)ncol * DIM;
    #pragma unroll
    for (int kc = 0; kc < DIM / 32; ++kc) {
        v16bf a = *(const v16bf*)&Hs[mrow * DIM + kc * 32 + khalf * 16];
        v16bf b = *(const v16bf*)&w1[kc * 32 + khalf * 16];
        acc2 = __builtin_amdgcn_wmma_f32_16x16x32_bf16(false, a, false, b, (short)0, acc2, false, false);
    }
    const float bc1 = b1[ncol];
    #pragma unroll
    for (int r = 0; r < 8; ++r) {
        const int m  = r + khalf * 8;
        const int rg = row0 + m;
        const int eg = idx_g[rg];
        const int es = idx_s[rg];
        float v = silu_f(acc2[r] + bc1) * m_pre[(size_t)eg * DIM + ncol];
        atomic_add_f32(&seg[(size_t)es * DIM + ncol], v);
    }
}

// ---------------- m = concat(h1[i], h1[j], rbf) : [E, 768] ----------------
__global__ void build_m_kernel(const float* __restrict__ h1, const float* __restrict__ rbf,
                               const int* __restrict__ edge_index, float* __restrict__ m)
{
    int b = blockIdx.x;
    int e = b / 3;
    int s = b - e * 3;
    int c = threadIdx.x;
    int jn = edge_index[e];            // edge_index[0]
    int in = edge_index[NEDGES + e];   // edge_index[1]
    float v;
    if (s == 0)      v = h1[(size_t)in * DIM + c];
    else if (s == 1) v = h1[(size_t)jn * DIM + c];
    else             v = rbf[(size_t)e * DIM + c];
    m[(size_t)e * 768 + s * DIM + c] = v;
}

// ---------------- h_agg[i[e]] += m3[e] ----------------
__global__ void scatter_edges_kernel(const float* __restrict__ m3,
                                     const int* __restrict__ edge_index,
                                     float* __restrict__ hagg)
{
    int e = blockIdx.x;
    int c = threadIdx.x;
    int in = edge_index[NEDGES + e];
    atomic_add_f32(&hagg[(size_t)in * DIM + c], m3[(size_t)e * DIM + c]);
}

// ---------------- y[n] = dot(y3[n,:], Wy) + by ----------------
__global__ __launch_bounds__(256) void y_out_kernel(const float* __restrict__ y3,
                                                    const float* __restrict__ Wy,
                                                    const float* __restrict__ by,
                                                    float* __restrict__ y)
{
    __shared__ float red[8];
    int n = blockIdx.x;
    int tid = threadIdx.x;
    float v = y3[(size_t)n * DIM + tid] * Wy[tid];
    #pragma unroll
    for (int off = 16; off > 0; off >>= 1) v += __shfl_down(v, off, 32);
    if ((tid & 31) == 0) red[tid >> 5] = v;
    __syncthreads();
    if (tid == 0) {
        float s = by[0];
        #pragma unroll
        for (int w = 0; w < 8; ++w) s += red[w];
        y[n] = s;
    }
}

// ================================ host side ================================

struct Ptrs {
    const float *h, *rbf, *sbf1, *sbf2;
    const int *idx_kj, *idx_ji1, *idx_jj, *idx_ji2, *edge_index;
    const float *Wh, *bh, *Wkj, *bkj, *Wji1, *bji1, *Wji2, *bji2, *Wjj, *bjj;
    const float *Ws10, *bs10, *Ws11, *bs11, *Ws20, *bs20, *Ws21, *bs21;
    const float *Wr1, *Wr2, *Wro;
    const float *Wres[3][4];            // {W0,b0,W1,b1} x res1..3
    const float *Wy0, *by0, *Wy1, *by1, *Wy2, *by2;
    const float *WY, *bY;
};

static void fill_ptrs(void* const* d, const int* sz, Ptrs& p) {
    auto F = [&](int i) { return (const float*)d[i]; };
    auto I = [&](int i) { return (const int*)d[i]; };
    if (sz[0] == NNODES * DIM) {
        // insertion order (setup_inputs dict order, params nested in insertion order)
        p.h = F(0); p.rbf = F(1); p.sbf1 = F(2); p.sbf2 = F(3);
        p.idx_kj = I(4); p.idx_ji1 = I(5); p.idx_jj = I(6); p.idx_ji2 = I(7);
        p.edge_index = I(8);
        p.Wh = F(9);  p.bh = F(10);
        p.Wkj = F(11); p.bkj = F(12);
        p.Wji1 = F(13); p.bji1 = F(14);
        p.Wji2 = F(15); p.bji2 = F(16);
        p.Wjj = F(17); p.bjj = F(18);
        p.Ws10 = F(19); p.bs10 = F(20); p.Ws11 = F(21); p.bs11 = F(22);
        p.Ws20 = F(23); p.bs20 = F(24); p.Ws21 = F(25); p.bs21 = F(26);
        p.Wr1 = F(27); p.Wr2 = F(28); p.Wro = F(29);
        for (int r = 0; r < 3; ++r)
            for (int q = 0; q < 4; ++q) p.Wres[r][q] = F(30 + r * 4 + q);
        p.Wy0 = F(42); p.by0 = F(43); p.Wy1 = F(44); p.by1 = F(45); p.Wy2 = F(46); p.by2 = F(47);
        p.WY = F(48); p.bY = F(49);
    } else {
        // jax pytree flatten (alphabetically sorted keys, recursively)
        p.edge_index = I(0); p.h = F(1);
        p.idx_ji1 = I(2); p.idx_ji2 = I(3); p.idx_jj = I(4); p.idx_kj = I(5);
        p.Wh = F(6); p.bh = F(7);
        p.Wr1 = F(8); p.Wr2 = F(9); p.Wro = F(10);
        p.Wji1 = F(11); p.bji1 = F(12);
        p.Wji2 = F(13); p.bji2 = F(14);
        p.Wjj = F(15); p.bjj = F(16);
        p.Wkj = F(17); p.bkj = F(18);
        p.Ws10 = F(19); p.bs10 = F(20); p.Ws11 = F(21); p.bs11 = F(22);
        p.Ws20 = F(23); p.bs20 = F(24); p.Ws21 = F(25); p.bs21 = F(26);
        for (int r = 0; r < 3; ++r)
            for (int q = 0; q < 4; ++q) p.Wres[r][q] = F(27 + r * 4 + q);
        p.WY = F(39); p.bY = F(40);
        p.Wy0 = F(41); p.by0 = F(42); p.Wy1 = F(43); p.by1 = F(44); p.Wy2 = F(45); p.by2 = F(46);
        p.rbf = F(47); p.sbf1 = F(48); p.sbf2 = F(49);
    }
}

static inline char* bump(char*& p, size_t bytes) {
    char* r = p;
    p += (bytes + 255) & ~(size_t)255;
    return r;
}

extern "C" void kernel_launch(void* const* d_in, const int* in_sizes, int n_in,
                              void* d_out, int out_size, void* d_ws, size_t ws_size,
                              hipStream_t stream)
{
    (void)n_in; (void)out_size; (void)ws_size;
    Ptrs P;
    fill_ptrs(d_in, in_sizes, P);

    // ---------------- workspace layout ----------------
    char* wsp = (char*)d_ws;
    auto abf = [&](size_t elems) { return (__bf16*)bump(wsp, elems * sizeof(__bf16)); };
    auto af  = [&](size_t elems) { return (float*)bump(wsp, elems * sizeof(float)); };

    __bf16* wt_h   = abf(256 * 256);
    __bf16* wt_kj  = abf(768 * 256);
    __bf16* wt_ji1 = abf(768 * 256);
    __bf16* wt_ji2 = abf(256 * 256);
    __bf16* wt_jj  = abf(256 * 256);
    __bf16* wt_s10 = abf(256 * 256);
    __bf16* wt_s11 = abf(256 * 256);
    __bf16* wt_s20 = abf(256 * 256);
    __bf16* wt_s21 = abf(256 * 256);
    __bf16* wt_r1  = abf(256 * 256);
    __bf16* wt_r2  = abf(256 * 256);
    __bf16* wt_ro  = abf(256 * 256);
    __bf16* wt_res[3][2];
    for (int r = 0; r < 3; ++r) { wt_res[r][0] = abf(256 * 256); wt_res[r][1] = abf(256 * 256); }
    __bf16* wt_y0  = abf(256 * 256);
    __bf16* wt_y1  = abf(256 * 256);
    __bf16* wt_y2  = abf(256 * 256);

    float* h1     = af((size_t)NNODES * DIM);
    float* hagg   = af((size_t)NNODES * DIM);
    float* nb1    = af((size_t)NNODES * DIM);
    float* nb2    = af((size_t)NNODES * DIM);
    float* nb3    = af((size_t)NNODES * DIM);
    float* buf_m  = af((size_t)NEDGES * 768);
    float* buf_r  = af((size_t)NEDGES * DIM);
    float* buf_p  = af((size_t)NEDGES * DIM);
    float* buf_sg = af((size_t)NEDGES * DIM);
    float* buf_m2 = af((size_t)NEDGES * DIM);

    float* h_out = (float*)d_out;
    float* y_out = h_out + (size_t)NNODES * DIM;

    // ---------------- weight prep (fp32 -> transposed bf16) ----------------
    struct WP { const float* W; __bf16* Wt; int K; };
    const WP wps[] = {
        {P.Wh, wt_h, 256}, {P.Wkj, wt_kj, 768}, {P.Wji1, wt_ji1, 768},
        {P.Wji2, wt_ji2, 256}, {P.Wjj, wt_jj, 256},
        {P.Ws10, wt_s10, 256}, {P.Ws11, wt_s11, 256},
        {P.Ws20, wt_s20, 256}, {P.Ws21, wt_s21, 256},
        {P.Wr1, wt_r1, 256}, {P.Wr2, wt_r2, 256}, {P.Wro, wt_ro, 256},
        {P.Wres[0][0], wt_res[0][0], 256}, {P.Wres[0][2], wt_res[0][1], 256},
        {P.Wres[1][0], wt_res[1][0], 256}, {P.Wres[1][2], wt_res[1][1], 256},
        {P.Wres[2][0], wt_res[2][0], 256}, {P.Wres[2][2], wt_res[2][1], 256},
        {P.Wy0, wt_y0, 256}, {P.Wy1, wt_y1, 256}, {P.Wy2, wt_y2, 256},
    };
    for (const WP& w : wps)
        prep_w_kernel<<<w.K, 256, 0, stream>>>(w.W, w.Wt, w.K);

    const int GN = NNODES / 16, GE = NEDGES / 16, GT = NTRI / 16;

    // ---------------- edge / triplet pipeline ----------------
    // h1 = silu(h @ Wh + bh)
    gemm_bf16_kernel<256, true><<<GN, 512, 0, stream>>>(P.h, wt_h, P.bh, nullptr, nullptr, h1);
    // m = concat(h1[i], h1[j], rbf)
    build_m_kernel<<<NEDGES * 3, 256, 0, stream>>>(h1, P.rbf, P.edge_index, buf_m);
    // rbf1 = rbf @ Wr1
    gemm_bf16_kernel<256, false><<<GE, 512, 0, stream>>>(P.rbf, wt_r1, nullptr, nullptr, nullptr, buf_r);
    // m_kj_pre = silu(m @ Wkj + b) * rbf1
    gemm_bf16_kernel<768, true><<<GE, 512, 0, stream>>>(buf_m, wt_kj, P.bkj, nullptr, buf_r, buf_p);
    // seg1 = segment_sum(m_kj_pre[idx_kj] * sbf1mlp, idx_ji_1)
    hipMemsetAsync(buf_sg, 0, (size_t)NEDGES * DIM * sizeof(float), stream);
    sbf_mlp_scatter_kernel<<<GT, 512, 0, stream>>>(P.sbf1, wt_s10, P.bs10, wt_s11, P.bs11,
                                                   buf_p, P.idx_kj, P.idx_ji1, buf_sg);
    // m2 = silu(m @ Wji1 + b) + seg1
    gemm_bf16_kernel<768, true><<<GE, 512, 0, stream>>>(buf_m, wt_ji1, P.bji1, buf_sg, nullptr, buf_m2);
    // rbf2 = rbf @ Wr2
    gemm_bf16_kernel<256, false><<<GE, 512, 0, stream>>>(P.rbf, wt_r2, nullptr, nullptr, nullptr, buf_r);
    // m_jj_pre = silu(m2 @ Wjj + b) * rbf2
    gemm_bf16_kernel<256, true><<<GE, 512, 0, stream>>>(buf_m2, wt_jj, P.bjj, nullptr, buf_r, buf_p);
    // seg2 = segment_sum(m_jj_pre[idx_jj] * sbf2mlp, idx_ji_2)
    hipMemsetAsync(buf_sg, 0, (size_t)NEDGES * DIM * sizeof(float), stream);
    sbf_mlp_scatter_kernel<<<GT, 512, 0, stream>>>(P.sbf2, wt_s20, P.bs20, wt_s21, P.bs21,
                                                   buf_p, P.idx_jj, P.idx_ji2, buf_sg);
    // rbf_out = rbf @ Wro
    gemm_bf16_kernel<256, false><<<GE, 512, 0, stream>>>(P.rbf, wt_ro, nullptr, nullptr, nullptr, buf_r);
    // m3 = (silu(m2 @ Wji2 + b) + seg2) * rbf_out
    gemm_bf16_kernel<256, true><<<GE, 512, 0, stream>>>(buf_m2, wt_ji2, P.bji2, buf_sg, buf_r, buf_p);
    // h_agg = segment_sum(m3, i)
    hipMemsetAsync(hagg, 0, (size_t)NNODES * DIM * sizeof(float), stream);
    scatter_edges_kernel<<<NEDGES, 256, 0, stream>>>(buf_p, P.edge_index, hagg);

    // ---------------- node pipeline ----------------
    // res1
    gemm_bf16_kernel<256, true><<<GN, 512, 0, stream>>>(hagg, wt_res[0][0], P.Wres[0][1], nullptr, nullptr, nb1);
    gemm_bf16_kernel<256, true><<<GN, 512, 0, stream>>>(nb1, wt_res[0][1], P.Wres[0][3], hagg, nullptr, nb2);
    // h_mlp + res_h
    gemm_bf16_kernel<256, true><<<GN, 512, 0, stream>>>(nb2, wt_h, P.bh, P.h, nullptr, nb3);
    // res2
    gemm_bf16_kernel<256, true><<<GN, 512, 0, stream>>>(nb3, wt_res[1][0], P.Wres[1][1], nullptr, nullptr, nb1);
    gemm_bf16_kernel<256, true><<<GN, 512, 0, stream>>>(nb1, wt_res[1][1], P.Wres[1][3], nb3, nullptr, nb2);
    // res3 -> h_out (d_out)
    gemm_bf16_kernel<256, true><<<GN, 512, 0, stream>>>(nb2, wt_res[2][0], P.Wres[2][1], nullptr, nullptr, nb1);
    gemm_bf16_kernel<256, true><<<GN, 512, 0, stream>>>(nb1, wt_res[2][1], P.Wres[2][3], nb2, nullptr, h_out);

    // y head
    gemm_bf16_kernel<256, true><<<GN, 512, 0, stream>>>(h_out, wt_y0, P.by0, nullptr, nullptr, nb1);
    gemm_bf16_kernel<256, true><<<GN, 512, 0, stream>>>(nb1, wt_y1, P.by1, nullptr, nullptr, nb2);
    gemm_bf16_kernel<256, true><<<GN, 512, 0, stream>>>(nb2, wt_y2, P.by2, nullptr, nullptr, nb3);
    y_out_kernel<<<NNODES, 256, 0, stream>>>(nb3, P.WY, P.bY, y_out);
}
